// WeightSparseTransformer_188978561465
// MI455X (gfx1250) — matrix operand order repaired and checked by reading, not compiled, not run on hardware
//
#include <hip/hip_runtime.h>
#include <hip/hip_bf16.h>
#include <math.h>

// Problem constants (must match reference)
#define B_   2
#define S_   1024
#define H_   512
#define NH_  8
#define HD_  64
#define L_   4
#define MLP_ 2048
#define OUT_ 128
#define KTOP 204            // int(1024 * (1.0 - 0.8)) = 204
#define EPS_ 1e-5f

typedef unsigned short u16;
typedef __bf16 bf16t;
typedef bf16t v16bf __attribute__((ext_vector_type(16)));
typedef float v8f   __attribute__((ext_vector_type(8)));
typedef u16   u16x8 __attribute__((ext_vector_type(8)));
typedef int   v4i_gcc __attribute__((vector_size(4 * sizeof(int))));

union Frag {
  u16x8 h[2];
  v16bf bf;
};

__device__ __forceinline__ u16 f2bf(float f) {
  unsigned u = __float_as_uint(f);
  u += 0x7FFFu + ((u >> 16) & 1u);   // round-to-nearest-even
  return (u16)(u >> 16);
}

// ---- CDNA5 async global->LDS copies (ASYNCcnt path), feature-guarded ------
#if defined(__HIP_DEVICE_COMPILE__) && __has_builtin(__builtin_amdgcn_global_load_async_to_lds_b128)
#define ASYNC_LDS 1
#else
#define ASYNC_LDS 0
#endif

#if ASYNC_LDS
// Builtin signature (from clang diagnostic): (int4 AS1*, int4 AS3*, Imm, Imm).
// generic->AS1 keeps the value; generic->AS3 is the low 32 bits of the generic
// address (ISA flat-aperture rule).
#define GLA_B128(gsrc, ldst)                                                    \
  __builtin_amdgcn_global_load_async_to_lds_b128(                               \
      (__attribute__((address_space(1))) v4i_gcc*)(uintptr_t)(const void*)(gsrc), \
      (__attribute__((address_space(3))) v4i_gcc*)(unsigned)(uintptr_t)(const void*)(ldst), \
      0, 0)
#if __has_builtin(__builtin_amdgcn_s_wait_asynccnt)
#define WAIT_ASYNC() __builtin_amdgcn_s_wait_asynccnt(0)
#else
#define WAIT_ASYNC() asm volatile("s_wait_asynccnt 0" ::: "memory")
#endif
#else
#define WAIT_ASYNC()
#endif

// ---------------------------------------------------------------------------
// Weight prep: Wt[n][k] = bf16(W[k][n]) once per launch; LDS-tiled transpose.
// All weight GEMMs then use contiguous 16B rows -> pure async-LDS staging.
// ---------------------------------------------------------------------------
__global__ __launch_bounds__(256)
void transpose_bf16(const float* __restrict__ W, u16* __restrict__ Wt,
                    int Kdim, int Ndim) {
  __shared__ u16 tile[32][33];
  const int n0 = blockIdx.x * 32, k0 = blockIdx.y * 32;
  const long long base = (long long)blockIdx.z * Kdim * Ndim;
  const int tx = threadIdx.x & 31, ty = threadIdx.x >> 5;   // 32 x 8
#pragma unroll
  for (int r = 0; r < 4; ++r) {
    const int k = ty + r * 8;
    tile[k][tx] = f2bf(W[base + (long long)(k0 + k) * Ndim + n0 + tx]);
  }
  __syncthreads();
#pragma unroll
  for (int r = 0; r < 4; ++r) {
    const int n = ty + r * 8;
    Wt[base + (long long)(n0 + n) * Kdim + k0 + tx] = tile[tx][n];
  }
}

// ---------------------------------------------------------------------------
// Embed: h[b,s,:] = x[b,s]*in_w[:] + in_b[:] + sinusoidal PE(s,:)
// ---------------------------------------------------------------------------
__global__ __launch_bounds__(256)
void embed_kernel(const float* __restrict__ x, const float* __restrict__ in_w,
                  const float* __restrict__ in_b, float* __restrict__ h) {
  long long i = (long long)blockIdx.x * 256 + threadIdx.x;   // over B*S*H
  int j = (int)(i % H_);
  long long bs = i / H_;
  int s = (int)(bs % S_);
  int pairbase = j & ~1;                                     // arange(0,H,2)
  float div = expf(-9.210340371976184f * (float)pairbase / (float)H_);
  float ang = (float)s * div;
  float pe = (j & 1) ? cosf(ang) : sinf(ang);
  h[i] = x[bs] * in_w[j] + in_b[j] + pe;
}

// ---------------------------------------------------------------------------
// LayerNorm over H=512, one wave (32 lanes x 16 elems) per token, bf16 out
// ---------------------------------------------------------------------------
__global__ __launch_bounds__(256)
void layernorm_bf16(const float* __restrict__ x, const float* __restrict__ g,
                    const float* __restrict__ b, u16* __restrict__ out) {
  const int wid = threadIdx.x >> 5, lane = threadIdx.x & 31;
  const int tok = blockIdx.x * 8 + wid;
  const float* row = x + (long long)tok * H_;
  float v[16];
  const float4* r4 = (const float4*)(row + lane * 16);
  float s = 0.f, ss = 0.f;
#pragma unroll
  for (int q = 0; q < 4; ++q) {
    float4 f = r4[q];
    v[q * 4 + 0] = f.x; v[q * 4 + 1] = f.y; v[q * 4 + 2] = f.z; v[q * 4 + 3] = f.w;
  }
#pragma unroll
  for (int e = 0; e < 16; ++e) { s += v[e]; ss += v[e] * v[e]; }
#pragma unroll
  for (int off = 16; off; off >>= 1) {
    s  += __shfl_xor(s,  off, 32);
    ss += __shfl_xor(ss, off, 32);
  }
  float mean = s * (1.0f / H_);
  float var  = ss * (1.0f / H_) - mean * mean;
  float rstd = rsqrtf(var + EPS_);
  u16* orow = out + (long long)tok * H_ + lane * 16;
#pragma unroll
  for (int e = 0; e < 16; ++e) {
    int idx = lane * 16 + e;
    orow[e] = f2bf((v[e] - mean) * rstd * g[idx] + b[idx]);
  }
}

// ---------------------------------------------------------------------------
// WMMA GEMM: C[M,N] = epi(alpha * A[M,K] @ B^T[N,K] + bias)
// 256 threads = 8 waves; wave tile (TM*16)x(TN*16) of v_wmma_f32_16x16x32_bf16;
// block tile 128 x BN; K-step 32; ping-pong LDS double buffer, software
// pipeline: async-LDS copies (or reg-staged loads) for the next tile issued
// before the current tile's WMMAs. B is ALWAYS bf16 row-major [N][K]
// (pre-transposed weights / K matrix / transposed V) -> contiguous async b128.
//   AF32: A source is f32 (attention probs; converted to bf16 in staging)
// Epilogue: 0=f32 store, 1=bf16 store, 2=exact-GELU+bf16, 3=f32 residual add,
//           4=bf16 store transposed per head (V path: vt[b][n][s])
// Batched over blockIdx.z with (b,head) stride pairs per operand.
// Fragment layouts per CDNA5 ISA 7.12.2.
// ---------------------------------------------------------------------------
template <int TM, int TN, int WAVES_M, bool AF32, int EPI>
__global__ __launch_bounds__(256)
void gemm_wmma(const void* __restrict__ Ap, int lda,
               const u16* __restrict__ Bp, int ldb,
               void* __restrict__ Cp, int ldc,
               const float* __restrict__ bias,
               int K, float alpha, int nh,
               long long sa1, long long sa2,
               long long sb1, long long sb2,
               long long sc1, long long sc2) {
  constexpr int BM = WAVES_M * TM * 16;
  constexpr int WAVES_N = 8 / WAVES_M;
  constexpr int BN = WAVES_N * TN * 16;
  static_assert(BM == 128, "A staging assumes BM==128");
  static_assert(BN == 64 || BN == 128, "B staging assumes BN in {64,128}");

  __shared__ __align__(16) u16 sA[2][BM][40];   // [m][k], padded stride
  __shared__ __align__(16) u16 sB[2][BN][40];   // [n][k], padded stride

  const int z  = blockIdx.z;
  const int zb = z / nh, zh = z % nh;
  const long long aoff = (long long)zb * sa1 + (long long)zh * sa2;
  const long long boff = (long long)zb * sb1 + (long long)zh * sb2;
  const long long coff = (long long)zb * sc1 + (long long)zh * sc2;

  const int tileM = blockIdx.x * BM;
  const int tileN = blockIdx.y * BN;
  const int t = threadIdx.x;
  const int wave = t >> 5, lane = t & 31;
  const int wm = wave % WAVES_M, wn = wave / WAVES_M;
  const int lr = lane & 15, hi2 = lane >> 4;

  v8f acc[TM][TN] = {};

  // staging assignments (256 threads)
  const int arow = t >> 1, ahalf = (t & 1) * 16;     // A: 16 elems/thread
  constexpr int BCH = (BN == 128) ? 16 : 8;          // B chunk per thread
  const int bn = t / (32 / BCH);
  const int bkb = (t % (32 / BCH)) * BCH;

  // pipeline staging registers (sync fallback / AF32 conversion)
  float4 regAf[4];
  uint4  regA[2];
  uint4  regB[2];

  auto gload = [&](int k0, int sel) {
    // ---- A tile (128 x 32) ----
    if constexpr (AF32) {
      const float* A = (const float*)Ap + aoff +
                       (long long)(tileM + arow) * lda + k0 + ahalf;
      const float4* A4 = (const float4*)A;
      regAf[0] = A4[0]; regAf[1] = A4[1]; regAf[2] = A4[2]; regAf[3] = A4[3];
      if (k0 + 32 < K) __builtin_prefetch((const void*)(A + 32), 0, 0);
    } else {
      const u16* A = (const u16*)Ap + aoff +
                     (long long)(tileM + arow) * lda + k0 + ahalf;
#if ASYNC_LDS
      GLA_B128(A,     &sA[sel][arow][ahalf]);
      GLA_B128(A + 8, &sA[sel][arow][ahalf + 8]);
#else
      regA[0] = *(const uint4*)A;
      regA[1] = *(const uint4*)(A + 8);
#endif
    }
    // ---- B tile (BN x 32), bf16 [N][K] rows ----
    const u16* Bq = Bp + boff + (long long)(tileN + bn) * ldb + k0 + bkb;
#if ASYNC_LDS
    GLA_B128(Bq, &sB[sel][bn][bkb]);
    if constexpr (BCH == 16) GLA_B128(Bq + 8, &sB[sel][bn][bkb + 8]);
#else
    regB[0] = *(const uint4*)Bq;
    if constexpr (BCH == 16) regB[1] = *(const uint4*)(Bq + 8);
#endif
  };

  auto lstore = [&](int sel) {
    if constexpr (AF32) {
      u16* d = &sA[sel][arow][ahalf];
#pragma unroll
      for (int q = 0; q < 4; ++q) {
        d[q * 4 + 0] = f2bf(regAf[q].x); d[q * 4 + 1] = f2bf(regAf[q].y);
        d[q * 4 + 2] = f2bf(regAf[q].z); d[q * 4 + 3] = f2bf(regAf[q].w);
      }
    }
#if !ASYNC_LDS
    if constexpr (!AF32) {
      *(uint4*)&sA[sel][arow][ahalf]     = regA[0];
      *(uint4*)&sA[sel][arow][ahalf + 8] = regA[1];
    }
    *(uint4*)&sB[sel][bn][bkb] = regB[0];
    if constexpr (BCH == 16) *(uint4*)&sB[sel][bn][bkb + 8] = regB[1];
#endif
  };

  // ---- software pipeline over K ----
  gload(0, 0);
  lstore(0);
  int cur = 0;
  for (int k0 = 0; k0 < K; k0 += 32) {
    WAIT_ASYNC();            // this wave's async tile commits (ASYNCcnt)
    __syncthreads();         // all waves' tile 'cur' visible
    const bool more = (k0 + 32) < K;
    if (more) gload(k0 + 32, cur ^ 1);   // overlap next-tile fetch with WMMAs

    Frag fa[TM], fb[TN];
#pragma unroll
    for (int i = 0; i < TM; ++i) {
      const int r = wm * (TM * 16) + i * 16 + lr;
      fa[i].h[0] = *(const u16x8*)&sA[cur][r][hi2 * 8];
      fa[i].h[1] = *(const u16x8*)&sA[cur][r][16 + hi2 * 8];
    }
#pragma unroll
    for (int j = 0; j < TN; ++j) {
      const int n = wn * (TN * 16) + j * 16 + lr;
      fb[j].h[0] = *(const u16x8*)&sB[cur][n][hi2 * 8];
      fb[j].h[1] = *(const u16x8*)&sB[cur][n][16 + hi2 * 8];
    }
#pragma unroll
    for (int i = 0; i < TM; ++i)
#pragma unroll
      for (int j = 0; j < TN; ++j)
        acc[i][j] = __builtin_amdgcn_wmma_f32_16x16x32_bf16(
            false, fa[i].bf, false, fb[j].bf, (short)0, acc[i][j], false, false);

    if (more) lstore(cur ^ 1);
    cur ^= 1;
  }

  // ---- epilogue: D layout: lane<16 -> M=r, lane>=16 -> M=8+r; N=lane&15 ----
#pragma unroll
  for (int i = 0; i < TM; ++i)
#pragma unroll
    for (int j = 0; j < TN; ++j)
#pragma unroll
      for (int r = 0; r < 8; ++r) {
        const int row = tileM + wm * (TM * 16) + i * 16 + hi2 * 8 + r;
        const int col = tileN + wn * (TN * 16) + j * 16 + lr;
        float v = alpha * acc[i][j][r];
        if (bias) v += bias[col];
        const long long cidx = coff + (long long)row * ldc + col;
        if (EPI == 0) {
          ((float*)Cp)[cidx] = v;
        } else if (EPI == 1) {
          ((u16*)Cp)[cidx] = f2bf(v);
        } else if (EPI == 2) {
          v = 0.5f * v * (1.0f + erff(v * 0.7071067811865476f));
          ((u16*)Cp)[cidx] = f2bf(v);
        } else if (EPI == 3) {
          ((float*)Cp)[cidx] += v;   // residual add, unique writer per element
        } else {
          // V path: vt[b][col][s] with b=row/S, s=row%S  (so attn@V's B operand
          // is row-major [N][K] -> async-capable)
          const long long tidx = (long long)col * S_ +
                                 (long long)(row >> 10) * ((long long)H_ * S_) +
                                 (row & (S_ - 1));
          ((u16*)Cp)[tidx] = f2bf(v);
        }
      }
}

// ---------------------------------------------------------------------------
// Exact top-k (k=204) threshold + masked softmax, in place on one 1024-row.
// Binary search on the order-preserving uint encoding -> exact k-th largest,
// keep scores >= kth (ties kept, matching the reference's where(s >= kth)).
// ---------------------------------------------------------------------------
__global__ __launch_bounds__(256)
void topk_softmax(float* __restrict__ scores) {
  __shared__ int   scnt[8];
  __shared__ float sred[8];
  float* row = scores + (long long)blockIdx.x * S_;
  const int t = threadIdx.x, wid = t >> 5, lane = t & 31;

  float4 xv = ((const float4*)row)[t];
  float x[4] = {xv.x, xv.y, xv.z, xv.w};
  unsigned u[4];
#pragma unroll
  for (int e = 0; e < 4; ++e) {
    unsigned bi = __float_as_uint(x[e]);
    u[e] = (bi & 0x80000000u) ? ~bi : (bi | 0x80000000u);
  }

  unsigned lo = 0u, hi = 0xFFFFFFFFu;   // invariant: count(u >= lo) >= k
  for (int it = 0; it < 32; ++it) {
    unsigned mid = lo + ((hi - lo) >> 1);
    if (mid == lo) break;               // uniform across block
    int c = (u[0] >= mid) + (u[1] >= mid) + (u[2] >= mid) + (u[3] >= mid);
#pragma unroll
    for (int off = 16; off; off >>= 1) c += __shfl_xor(c, off, 32);
    if (lane == 0) scnt[wid] = c;
    __syncthreads();
    int tot = 0;
#pragma unroll
    for (int w = 0; w < 8; ++w) tot += scnt[w];
    __syncthreads();
    if (tot >= KTOP) lo = mid; else hi = mid;
  }
  const unsigned thr = lo;              // == k-th largest (order-encoded)

  float m = -3.0e38f;
#pragma unroll
  for (int e = 0; e < 4; ++e) if (u[e] >= thr) m = fmaxf(m, x[e]);
#pragma unroll
  for (int off = 16; off; off >>= 1) m = fmaxf(m, __shfl_xor(m, off, 32));
  if (lane == 0) sred[wid] = m;
  __syncthreads();
  float mm = sred[0];
#pragma unroll
  for (int w = 1; w < 8; ++w) mm = fmaxf(mm, sred[w]);
  __syncthreads();

  float y[4]; float sum = 0.f;
#pragma unroll
  for (int e = 0; e < 4; ++e) {
    y[e] = (u[e] >= thr) ? expf(x[e] - mm) : 0.f;
    sum += y[e];
  }
#pragma unroll
  for (int off = 16; off; off >>= 1) sum += __shfl_xor(sum, off, 32);
  if (lane == 0) sred[wid] = sum;
  __syncthreads();
  float ts = 0.f;
#pragma unroll
  for (int w = 0; w < 8; ++w) ts += sred[w];
  float inv = 1.0f / ts;
  float4 yo = {y[0] * inv, y[1] * inv, y[2] * inv, y[3] * inv};
  ((float4*)row)[t] = yo;
}

// ---------------------------------------------------------------------------
extern "C" void kernel_launch(void* const* d_in, const int* in_sizes, int n_in,
                              void* d_out, int out_size, void* d_ws, size_t ws_size,
                              hipStream_t stream) {
  const float* x     = (const float*)d_in[0];
  const float* in_w  = (const float*)d_in[1];
  const float* in_b  = (const float*)d_in[2];
  const float* ln1_g = (const float*)d_in[3];
  const float* ln1_b = (const float*)d_in[4];
  const float* qw    = (const float*)d_in[5];
  const float* qb    = (const float*)d_in[6];
  const float* kw    = (const float*)d_in[7];
  const float* kb    = (const float*)d_in[8];
  const float* vw    = (const float*)d_in[9];
  const float* vb    = (const float*)d_in[10];
  const float* ow    = (const float*)d_in[11];
  const float* ob    = (const float*)d_in[12];
  const float* ln2_g = (const float*)d_in[13];
  const float* ln2_b = (const float*)d_in[14];
  const float* w1    = (const float*)d_in[15];
  const float* b1    = (const float*)d_in[16];
  const float* w2    = (const float*)d_in[17];
  const float* b2    = (const float*)d_in[18];
  const float* fn_g  = (const float*)d_in[19];
  const float* fn_b  = (const float*)d_in[20];
  const float* out_w = (const float*)d_in[21];
  const float* out_b = (const float*)d_in[22];

  char* wsp = (char*)d_ws;
  auto take = [&](size_t bytes) -> void* {
    void* p = (void*)wsp;
    wsp += (bytes + 255) & ~(size_t)255;
    return p;
  };
  const int M = B_ * S_;                                   // 2048 tokens
  float* h     = (float*)take((size_t)M * H_ * 4);         // residual stream
  u16*   nbuf  = (u16*)  take((size_t)M * H_ * 2);         // LN out (bf16)
  u16*   qbuf  = (u16*)  take((size_t)M * H_ * 2);
  u16*   kbuf  = (u16*)  take((size_t)M * H_ * 2);
  u16*   vtbuf = (u16*)  take((size_t)B_ * H_ * S_ * 2);   // V transposed/head
  u16*   obuf  = (u16*)  take((size_t)M * H_ * 2);
  u16*   m1buf = (u16*)  take((size_t)M * MLP_ * 2);
  // pre-transposed bf16 weights Wt[N][K]
  u16*   qwt   = (u16*)  take((size_t)L_ * H_ * H_ * 2);
  u16*   kwt   = (u16*)  take((size_t)L_ * H_ * H_ * 2);
  u16*   vwt   = (u16*)  take((size_t)L_ * H_ * H_ * 2);
  u16*   owt   = (u16*)  take((size_t)L_ * H_ * H_ * 2);
  u16*   w1t   = (u16*)  take((size_t)L_ * H_ * MLP_ * 2);
  u16*   w2t   = (u16*)  take((size_t)L_ * MLP_ * H_ * 2);
  u16*   outwt = (u16*)  take((size_t)H_ * OUT_ * 2);

  float* outp = (float*)d_out;                             // [B,S,OUT]
  float* attnbase = (float*)d_out + (size_t)B_ * S_ * OUT_;

  // ---- one-time weight transpose+convert (L2-resident thereafter) ----
  transpose_bf16<<<dim3(H_/32,   H_/32,   L_), 256, 0, stream>>>(qw,  qwt,  H_,   H_);
  transpose_bf16<<<dim3(H_/32,   H_/32,   L_), 256, 0, stream>>>(kw,  kwt,  H_,   H_);
  transpose_bf16<<<dim3(H_/32,   H_/32,   L_), 256, 0, stream>>>(vw,  vwt,  H_,   H_);
  transpose_bf16<<<dim3(H_/32,   H_/32,   L_), 256, 0, stream>>>(ow,  owt,  H_,   H_);
  transpose_bf16<<<dim3(MLP_/32, H_/32,   L_), 256, 0, stream>>>(w1,  w1t,  H_,   MLP_);
  transpose_bf16<<<dim3(H_/32,   MLP_/32, L_), 256, 0, stream>>>(w2,  w2t,  MLP_, H_);
  transpose_bf16<<<dim3(OUT_/32, H_/32,   1 ), 256, 0, stream>>>(out_w, outwt, H_, OUT_);

  embed_kernel<<<dim3((M * H_) / 256), 256, 0, stream>>>(x, in_w, in_b, h);

  for (int l = 0; l < L_; ++l) {
    layernorm_bf16<<<M / 8, 256, 0, stream>>>(h, ln1_g + l * H_, ln1_b + l * H_, nbuf);

    // QKV projections: [2048,512] @ Wt[512,512]^T -> bf16 (V stored transposed)
    gemm_wmma<4, 2, 2, false, 1><<<dim3(M/128, H_/128, 1), 256, 0, stream>>>(
        nbuf, H_, qwt + (size_t)l*H_*H_, H_, qbuf, H_, qb + l*H_,
        H_, 1.0f, 1, 0,0, 0,0, 0,0);
    gemm_wmma<4, 2, 2, false, 1><<<dim3(M/128, H_/128, 1), 256, 0, stream>>>(
        nbuf, H_, kwt + (size_t)l*H_*H_, H_, kbuf, H_, kb + l*H_,
        H_, 1.0f, 1, 0,0, 0,0, 0,0);
    gemm_wmma<4, 2, 2, false, 4><<<dim3(M/128, H_/128, 1), 256, 0, stream>>>(
        nbuf, H_, vwt + (size_t)l*H_*H_, H_, vtbuf, S_, vb + l*H_,
        H_, 1.0f, 1, 0,0, 0,0, 0,0);

    float* attn_l = attnbase + (size_t)l * B_ * NH_ * S_ * S_;

    // scores = scale * Q @ K^T per (b, head) -> directly into d_out attn slab
    gemm_wmma<4, 2, 2, false, 0><<<dim3(S_/128, S_/128, B_*NH_), 256, 0, stream>>>(
        qbuf, H_, kbuf, H_, attn_l, S_, nullptr,
        HD_, 0.125f, NH_,
        (long long)S_*H_, (long long)HD_,
        (long long)S_*H_, (long long)HD_,
        (long long)NH_*S_*S_, (long long)S_*S_);

    // exact top-k mask + softmax, in place
    topk_softmax<<<B_ * NH_ * S_, 256, 0, stream>>>(attn_l);

    // O_head = P @ V per (b, head): B = vt[b][h*HD+d][s] rows -> async staging
    gemm_wmma<2, 2, 4, true, 1><<<dim3(S_/128, HD_/64, B_*NH_), 256, 0, stream>>>(
        attn_l, S_, vtbuf, S_, obuf, H_, nullptr,
        S_, 1.0f, NH_,
        (long long)NH_*S_*S_, (long long)S_*S_,
        (long long)H_*S_, (long long)HD_*S_,
        (long long)S_*H_, (long long)HD_);

    // h += O @ Wo + bo
    gemm_wmma<4, 2, 2, false, 3><<<dim3(M/128, H_/128, 1), 256, 0, stream>>>(
        obuf, H_, owt + (size_t)l*H_*H_, H_, h, H_, ob + l*H_,
        H_, 1.0f, 1, 0,0, 0,0, 0,0);

    layernorm_bf16<<<M / 8, 256, 0, stream>>>(h, ln2_g + l * H_, ln2_b + l * H_, nbuf);

    // m1 = gelu(n @ W1 + b1)   W1t is [2048][512]
    gemm_wmma<4, 2, 2, false, 2><<<dim3(M/128, MLP_/128, 1), 256, 0, stream>>>(
        nbuf, H_, w1t + (size_t)l*H_*MLP_, H_, m1buf, MLP_, b1 + l*MLP_,
        H_, 1.0f, 1, 0,0, 0,0, 0,0);

    // h += m1 @ W2 + b2        W2t is [512][2048]
    gemm_wmma<4, 2, 2, false, 3><<<dim3(M/128, H_/128, 1), 256, 0, stream>>>(
        m1buf, MLP_, w2t + (size_t)l*MLP_*H_, MLP_, h, H_, b2 + l*H_,
        MLP_, 1.0f, 1, 0,0, 0,0, 0,0);
  }

  layernorm_bf16<<<M / 8, 256, 0, stream>>>(h, fn_g, fn_b, nbuf);

  // out = LN(h) @ out_w + out_b -> f32 [2048,128]   out_wt is [128][512]
  gemm_wmma<4, 2, 2, false, 0><<<dim3(M/128, OUT_/128, 1), 256, 0, stream>>>(
      nbuf, H_, outwt, H_, outp, OUT_, out_b,
      H_, 1.0f, 1, 0,0, 0,0, 0,0);
}
